// TripletLoss_10462540333670
// MI455X (gfx1250) — compile-verified, matrix operand order checked by reading, not run on hardware
//
#include <hip/hip_runtime.h>

#define DCOLS 512
#define WAVES_PER_BLOCK 4
#define JTILES 4   // 64 columns of the Gram matrix per j-group
#define NSPLIT 8   // j-range split factor (parallelism: N/16 * NSPLIT waves)

static constexpr float kEps    = 1e-6f;
static constexpr float kMargin = 0.2f;

typedef __attribute__((ext_vector_type(16))) __bf16   v16bf;
typedef __attribute__((ext_vector_type(8)))  float    v8f;
typedef __attribute__((ext_vector_type(4)))  unsigned u32x4;

union Frag { v16bf v; u32x4 q[2]; };

__device__ __forceinline__ unsigned short f32_to_bf16_rne(float f) {
  unsigned u = __float_as_uint(f);
  u += 0x7fffu + ((u >> 16) & 1u);   // round-to-nearest-even
  return (unsigned short)(u >> 16);
}
__device__ __forceinline__ float bf16u_to_f32(unsigned short h) {
  return __uint_as_float(((unsigned)h) << 16);
}

// ---------------------------------------------------------------------------
// Kernel 1: per-row sq = sum(a^2), s = sum(a), d_ap = ||a - p + eps||,
//           hi/lo bf16 split of the anchor matrix, and rowmin init (+inf).
// One 256-thread block per row (D = 512 -> 2 elements/thread).
// ---------------------------------------------------------------------------
__global__ __launch_bounds__(256)
void prep_kernel(const float* __restrict__ A, const float* __restrict__ P,
                 unsigned short* __restrict__ Ahi, unsigned short* __restrict__ Alo,
                 float* __restrict__ sq, float* __restrict__ s,
                 float* __restrict__ dap, unsigned* __restrict__ rowminBits) {
  const int row = blockIdx.x;
  const int t   = threadIdx.x;
  const float* a = A + (size_t)row * DCOLS;
  const float* p = P + (size_t)row * DCOLS;

  float sqv = 0.f, sv = 0.f, dv = 0.f;
  for (int e = t; e < DCOLS; e += 256) {
    float av = a[e];
    float pv = p[e];
    sqv = fmaf(av, av, sqv);
    sv += av;
    float diff = av - pv + kEps;
    dv = fmaf(diff, diff, dv);
    unsigned short hi = f32_to_bf16_rne(av);
    unsigned short lo = f32_to_bf16_rne(av - bf16u_to_f32(hi));
    Ahi[(size_t)row * DCOLS + e] = hi;
    Alo[(size_t)row * DCOLS + e] = lo;
  }

  __shared__ float r0[256], r1[256], r2[256];
  r0[t] = sqv; r1[t] = sv; r2[t] = dv;
  __syncthreads();
  for (int st = 128; st > 0; st >>= 1) {
    if (t < st) { r0[t] += r0[t + st]; r1[t] += r1[t + st]; r2[t] += r2[t + st]; }
    __syncthreads();
  }
  if (t == 0) {
    sq[row]  = r0[0];
    s[row]   = r1[0];
    dap[row] = sqrtf(r2[0]);
    rowminBits[row] = 0x7f800000u;   // +inf, re-initialized every call
  }
}

// ---------------------------------------------------------------------------
// 16x32 bf16 fragment loader (ISA 7.12.2 layout):
//   lanes 0-15 hold matrix row (lane), lanes 16-31 the same rows;
//   half = lane>>4 selects K = k+8*half .. +7 (VGPRs 0-3)
//   and K = k+16+8*half .. +7 (VGPRs 4-7). Two b128 loads per lane.
// For B = A^T this same pattern (row index = output column) gives the
// 32x16 B fragment, so the Gram matrix needs no transpose loads.
// ---------------------------------------------------------------------------
__device__ __forceinline__ v16bf load_frag(const unsigned short* __restrict__ base,
                                           int row, int k, int half) {
  Frag f;
  const unsigned short* p = base + (size_t)row * DCOLS + k + half * 8;
  f.q[0] = *(const u32x4*)(p);        // K rel 0..7   (+8*half)
  f.q[1] = *(const u32x4*)(p + 16);   // K rel 16..23 (+8*half)
  return f.v;
}

// ---------------------------------------------------------------------------
// Kernel 2: per-row hardest-negative squared distance via bf16x2 WMMA Gram.
// Each wave owns 16 rows and one of NSPLIT column chunks; it sweeps the chunk
// in groups of 64 columns (4 16x16 accumulator tiles), applies the d2
// epilogue, keeps a running per-lane min, min-reduces across each 16-lane
// half, and merges into the global row minima with a bitwise atomicMin
// (valid: all candidates are non-negative floats).
// ---------------------------------------------------------------------------
__global__ __launch_bounds__(32 * WAVES_PER_BLOCK)
void gram_rowmin_kernel(const unsigned short* __restrict__ Ahi,
                        const unsigned short* __restrict__ Alo,
                        const float* __restrict__ sq, const float* __restrict__ s,
                        unsigned* __restrict__ rowminBits, int N) {
  const int lane = threadIdx.x & 31;
  const int wave = threadIdx.x >> 5;
  const int half = lane >> 4;
  const int nloc = lane & 15;
  const int rowBase = (blockIdx.x * WAVES_PER_BLOCK + wave) * 16;
  const int arow = rowBase + nloc;   // A-fragment row for this lane

  const int jChunk = N / NSPLIT;
  const int jBegin = blockIdx.y * jChunk;
  const int jEnd   = jBegin + jChunk;

  // Per-lane row constants: C/D tile layout puts row M = v + 8*half in VGPR v.
  float rsq[8], rs[8];
#pragma unroll
  for (int v = 0; v < 8; ++v) {
    int i = rowBase + v + 8 * half;
    rsq[v] = sq[i];
    rs[v]  = s[i];
  }

  float rm[8];
#pragma unroll
  for (int v = 0; v < 8; ++v) rm[v] = __uint_as_float(0x7f800000u);  // +inf

  const float twoEps = 2.f * kEps;
  const float dEps2  = (float)DCOLS * kEps * kEps;

  for (int jBase = jBegin; jBase < jEnd; jBase += 16 * JTILES) {
    v8f acc[JTILES];
#pragma unroll
    for (int tt = 0; tt < JTILES; ++tt)
#pragma unroll
      for (int v = 0; v < 8; ++v) acc[tt][v] = 0.f;

    for (int k = 0; k < DCOLS; k += 32) {
      v16bf a_h = load_frag(Ahi, arow, k, half);
      v16bf a_l = load_frag(Alo, arow, k, half);
#pragma unroll
      for (int tt = 0; tt < JTILES; ++tt) {
        const int brow = jBase + tt * 16 + nloc;
        v16bf b_h = load_frag(Ahi, brow, k, half);
        v16bf b_l = load_frag(Alo, brow, k, half);
        // gram ~= hi*hi + hi*lo + lo*hi  (fp32 accumulate; lo*lo ~ 2^-16 rel, dropped)
        acc[tt] = __builtin_amdgcn_wmma_f32_16x16x32_bf16(
            false, a_h, false, b_h, (short)0, acc[tt], false, false);
        acc[tt] = __builtin_amdgcn_wmma_f32_16x16x32_bf16(
            false, a_h, false, b_l, (short)0, acc[tt], false, false);
        acc[tt] = __builtin_amdgcn_wmma_f32_16x16x32_bf16(
            false, a_l, false, b_h, (short)0, acc[tt], false, false);
      }
    }

    // Epilogue: d2 = sq_i + sq_j - 2 g + 2 eps (s_i - s_j) + D eps^2
#pragma unroll
    for (int tt = 0; tt < JTILES; ++tt) {
      const int j = jBase + tt * 16 + nloc;
      const float csq = sq[j];
      const float cs  = s[j];
#pragma unroll
      for (int v = 0; v < 8; ++v) {
        const int i = rowBase + v + 8 * half;
        float d2 = rsq[v] + csq - 2.f * acc[tt][v] + twoEps * (rs[v] - cs) + dEps2;
        d2 = fmaxf(d2, 0.f);
        d2 = (i == j) ? __uint_as_float(0x7f800000u) : d2;   // mask diagonal
        rm[v] = fminf(rm[v], d2);
      }
    }
  }

  // Cross-lane min within each 16-lane half (rows v and v+8 of this wave's
  // tile), then merge with the global per-row minimum. IEEE ordering of
  // non-negative floats matches unsigned ordering, so atomicMin on the bit
  // pattern computes the float min exactly (and order-independently).
#pragma unroll
  for (int v = 0; v < 8; ++v) {
    float m = rm[v];
#pragma unroll
    for (int off = 8; off > 0; off >>= 1)
      m = fminf(m, __shfl_xor(m, off, 16));
    if (nloc == 0)
      atomicMin(&rowminBits[rowBase + v + 8 * half], __float_as_uint(m));
  }
}

// ---------------------------------------------------------------------------
// Kernel 3: loss = mean(relu(d_ap - sqrt(rowmin) + margin))
// ---------------------------------------------------------------------------
__global__ __launch_bounds__(1024)
void loss_kernel(const float* __restrict__ dap, const float* __restrict__ rowmin,
                 float* __restrict__ out, int N) {
  const int t = threadIdx.x;
  float acc = 0.f;
  for (int i = t; i < N; i += 1024) {
    float dan = sqrtf(rowmin[i]);
    acc += fmaxf(dap[i] - dan + kMargin, 0.f);
  }
  __shared__ float red[1024];
  red[t] = acc;
  __syncthreads();
  for (int st = 512; st > 0; st >>= 1) {
    if (t < st) red[t] += red[t + st];
    __syncthreads();
  }
  if (t == 0) out[0] = red[0] / (float)N;
}

// ---------------------------------------------------------------------------
extern "C" void kernel_launch(void* const* d_in, const int* in_sizes, int n_in,
                              void* d_out, int out_size, void* d_ws, size_t ws_size,
                              hipStream_t stream) {
  (void)n_in; (void)out_size; (void)ws_size;
  const float* anchor   = (const float*)d_in[0];
  const float* positive = (const float*)d_in[1];
  float* out = (float*)d_out;
  const int N = in_sizes[0] / DCOLS;   // 8192 for the reference shapes

  // Workspace layout (~16.13 MB): bf16 hi | bf16 lo | sq | s | d_ap | rowmin
  char* ws = (char*)d_ws;
  const size_t bfBytes = (size_t)N * DCOLS * sizeof(unsigned short);
  unsigned short* Ahi = (unsigned short*)ws;
  unsigned short* Alo = (unsigned short*)(ws + bfBytes);
  float* sq     = (float*)(ws + 2 * bfBytes);
  float* s      = sq + N;
  float* dap    = s + N;
  unsigned* rowminBits = (unsigned*)(dap + N);

  prep_kernel<<<N, 256, 0, stream>>>(anchor, positive, Ahi, Alo, sq, s, dap,
                                     rowminBits);

  const int rowsPerBlock = 16 * WAVES_PER_BLOCK;
  dim3 grid(N / rowsPerBlock, NSPLIT);
  gram_rowmin_kernel<<<grid, 32 * WAVES_PER_BLOCK, 0, stream>>>(
      Ahi, Alo, sq, s, rowminBits, N);

  loss_kernel<<<1, 1024, 0, stream>>>(dap, (const float*)rowminBits, out, N);
}
